// DeformableDETREncoder_52733608460649
// MI455X (gfx1250) — matrix operand order repaired and checked by reading, not compiled, not run on hardware
//
#include <hip/hip_runtime.h>
#include <hip/hip_bf16.h>
#include <cstdint>
#include <cstddef>

// ---------------- problem constants (from reference) ----------------
#define NLAYERS 6
#define DM      256
#define DFFN    1024
#define NHEADS  8
#define NLEV    4
#define NPTS    4
#define HD      32
#define BB      2
#define QK      13294            // 100*100 + 50*50 + 25*25 + 13*13
#define MROWS   (BB * QK)        // 26588

typedef __attribute__((ext_vector_type(16))) _Float16 v16h;
typedef __attribute__((ext_vector_type(8)))  _Float16 v8h;
typedef __attribute__((ext_vector_type(8)))  float    v8f;
typedef __attribute__((ext_vector_type(4)))  float    v4f;

// ==================================================================
// x = concat_l( src_l (B,256,HW) transposed ) + pos  (pos broadcast over B)
// writes fp32 master copy AND f16 mirror for the WMMA GEMMs
// ==================================================================
__global__ void build_x_kernel(const float* __restrict__ src,
                               const float* __restrict__ pos,
                               float* __restrict__ x,
                               _Float16* __restrict__ xh, int HW, int start) {
  int idx = blockIdx.x * blockDim.x + threadIdx.x;
  int total = BB * HW * DM;
  if (idx >= total) return;
  int d = idx % DM;
  int i = (idx / DM) % HW;
  int b = idx / (DM * HW);
  float v = src[((size_t)b * DM + d) * HW + i] + pos[(size_t)i * DM + d];
  size_t o = ((size_t)b * QK + start + i) * DM + d;
  x[o]  = v;
  xh[o] = (_Float16)v;
}

// ==================================================================
// Wt[n*K + k] = (f16) W[k*N + n]   (transpose + convert, per layer)
// ==================================================================
__global__ void wconv_kernel(const float* __restrict__ W,
                             _Float16* __restrict__ Wt, int K, int N) {
  int idx = blockIdx.x * blockDim.x + threadIdx.x;
  if (idx >= K * N) return;
  int k = idx % K;
  int n = idx / K;
  Wt[(size_t)n * K + k] = (_Float16)W[(size_t)k * N + n];
}

// ==================================================================
// WMMA GEMM:  C(M,N) = A(M,K) * Wt(N,K)^T + bias [, relu]
// A f16 row-major, Wt f16 row-major (N x K).
// Output: f32 (Cf32) or f16 (Cf16) -- exactly one is non-null.
// Block = 128 threads = 4 waves (2x2), block tile 64x64, wave tile 32x32.
// ==================================================================
__device__ inline v16h load_h_frag(const _Float16* p, int gap) {
  // fragment = 8 contiguous f16 at p, then 8 contiguous f16 at p+gap
  // A (16x32): gap=16 (K groups half*8 and 16+half*8)
  // B (32x16): gap=16 (K groups half*8 and 16+half*8)
  v8h lo = *(const v8h*)(p);
  v8h hi = *(const v8h*)(p + gap);
  v16h r;
#pragma unroll
  for (int i = 0; i < 8; ++i) {
    r[i]     = lo[i];
    r[8 + i] = hi[i];
  }
  return r;
}

__device__ inline void store_tile(const v8f& c, int mBase, int nBase, int half,
                                  int l16, float* __restrict__ Cf32,
                                  _Float16* __restrict__ Cf16,
                                  const float* __restrict__ bias, int M, int N,
                                  int relu) {
  int col = nBase + l16;
  float bval = bias[col];
#pragma unroll
  for (int r = 0; r < 8; ++r) {
    int row = mBase + 8 * half + r;
    if (row < M) {
      float v = c[r] + bval;
      if (relu) v = fmaxf(v, 0.0f);
      if (Cf16) Cf16[(size_t)row * N + col] = (_Float16)v;
      else      Cf32[(size_t)row * N + col] = v;
    }
  }
}

__global__ __launch_bounds__(128) void gemm_wmma_kernel(
    const _Float16* __restrict__ A, const _Float16* __restrict__ Wt,
    const float* __restrict__ bias, float* __restrict__ Cf32,
    _Float16* __restrict__ Cf16, int M, int N, int K, int relu) {
  const int lane  = threadIdx.x & 31;
  const int wave  = threadIdx.x >> 5;
  const int waveM = wave >> 1;
  const int waveN = wave & 1;
  const int l16   = lane & 15;
  const int half  = lane >> 4;

  const int m0 = blockIdx.y * 64 + waveM * 32;
  const int n0 = blockIdx.x * 64 + waveN * 32;

  // clamped row indices so every lane loads valid memory (EXEC stays all-1s)
  int rA0 = m0 + l16;        if (rA0 > M - 1) rA0 = M - 1;
  int rA1 = m0 + 16 + l16;   if (rA1 > M - 1) rA1 = M - 1;
  const int nB0 = n0 + l16;
  const int nB1 = n0 + 16 + l16;

  const _Float16* pa0 = A  + (size_t)rA0 * K + half * 8;
  const _Float16* pa1 = A  + (size_t)rA1 * K + half * 8;
  const _Float16* pb0 = Wt + (size_t)nB0 * K + half * 8;
  const _Float16* pb1 = Wt + (size_t)nB1 * K + half * 8;

  v8f c00 = {}, c01 = {}, c10 = {}, c11 = {};

#pragma unroll 2
  for (int kb = 0; kb < K; kb += 32) {
    if (kb + 32 < K) {  // uniform guard; speculative prefetch of next K-block
      __builtin_prefetch(pa0 + kb + 32, 0, 1);
      __builtin_prefetch(pa1 + kb + 32, 0, 1);
    }
    v16h a0 = load_h_frag(pa0 + kb, 16);
    v16h a1 = load_h_frag(pa1 + kb, 16);
    v16h b0 = load_h_frag(pb0 + kb, 16);
    v16h b1 = load_h_frag(pb1 + kb, 16);
    c00 = __builtin_amdgcn_wmma_f32_16x16x32_f16(false, a0, false, b0, (short)0, c00, false, false);
    c01 = __builtin_amdgcn_wmma_f32_16x16x32_f16(false, a0, false, b1, (short)0, c01, false, false);
    c10 = __builtin_amdgcn_wmma_f32_16x16x32_f16(false, a1, false, b0, (short)0, c10, false, false);
    c11 = __builtin_amdgcn_wmma_f32_16x16x32_f16(false, a1, false, b1, (short)0, c11, false, false);
  }

  store_tile(c00, m0,      n0,      half, l16, Cf32, Cf16, bias, M, N, relu);
  store_tile(c01, m0,      n0 + 16, half, l16, Cf32, Cf16, bias, M, N, relu);
  store_tile(c10, m0 + 16, n0,      half, l16, Cf32, Cf16, bias, M, N, relu);
  store_tile(c11, m0 + 16, n0 + 16, half, l16, Cf32, Cf16, bias, M, N, relu);
}

// ==================================================================
// Softmax over 16 contiguous values per (b,q,h)
// ==================================================================
__global__ void softmax16_kernel(float* __restrict__ aw, int rows) {
  int idx = blockIdx.x * blockDim.x + threadIdx.x;
  if (idx >= rows) return;
  float* p = aw + (size_t)idx * 16;
  float mx = p[0];
#pragma unroll
  for (int j = 1; j < 16; ++j) mx = fmaxf(mx, p[j]);
  float e[16];
  float s = 0.f;
#pragma unroll
  for (int j = 0; j < 16; ++j) {
    e[j] = expf(p[j] - mx);
    s += e[j];
  }
  float inv = 1.0f / s;
#pragma unroll
  for (int j = 0; j < 16; ++j) p[j] = e[j] * inv;
}

// ==================================================================
// Deformable attention sampling: one block per (b,q); thread = (h,d)
// out is written in f16 (it feeds only the Wo WMMA GEMM)
// ==================================================================
__device__ inline float samp_tap(const float* __restrict__ val, size_t vrow0,
                                 int st, int W, int H, int xi, int yi, int col) {
  if (xi < 0 || xi >= W || yi < 0 || yi >= H) return 0.0f;
  return val[(vrow0 + st + (size_t)yi * W + xi) * DM + col];
}

__global__ __launch_bounds__(256) void msda_sample_kernel(
    const float* __restrict__ val, const float* __restrict__ off,
    const float* __restrict__ aw, const float* __restrict__ ref,
    _Float16* __restrict__ outh) {
  const int row = blockIdx.x;  // b*QK + q
  const int t = threadIdx.x;
  const int d = t & 31;
  const int h = t >> 5;
  const int b = row / QK;

  const float* offp = off + (size_t)row * DM;
  const float* awp  = aw  + (size_t)row * (NHEADS * 16);
  const float* refp = ref + (size_t)row * (NLEV * 2);
  const size_t vrow0 = (size_t)b * QK;
  const int col = h * HD + d;

  const int Hs[4] = {100, 50, 25, 13};
  const int Ws[4] = {100, 50, 25, 13};
  const int Ss[4] = {0, 10000, 12500, 13125};

  float acc = 0.0f;
#pragma unroll
  for (int l = 0; l < NLEV; ++l) {
    const int H = Hs[l], W = Ws[l], st = Ss[l];
    const float rx = refp[l * 2 + 0];
    const float ry = refp[l * 2 + 1];
#pragma unroll
    for (int p = 0; p < NPTS; ++p) {
      const int oi = ((h * NLEV + l) * NPTS + p) * 2;
      // norm order is (W, H) in the reference
      float lx = rx + offp[oi + 0] / (float)W;
      float ly = ry + offp[oi + 1] / (float)H;
      float xf = lx * (float)W - 0.5f;
      float yf = ly * (float)H - 0.5f;
      float x0 = floorf(xf), y0 = floorf(yf);
      float fx = xf - x0, fy = yf - y0;
      int xi = (int)x0, yi = (int)y0;
      float a = awp[h * 16 + l * 4 + p];
      float v00 = samp_tap(val, vrow0, st, W, H, xi,     yi,     col);
      float v10 = samp_tap(val, vrow0, st, W, H, xi + 1, yi,     col);
      float v01 = samp_tap(val, vrow0, st, W, H, xi,     yi + 1, col);
      float v11 = samp_tap(val, vrow0, st, W, H, xi + 1, yi + 1, col);
      acc += a * ((v00 * (1.f - fx) + v10 * fx) * (1.f - fy) +
                  (v01 * (1.f - fx) + v11 * fx) * fy);
    }
  }
  outh[(size_t)row * DM + col] = (_Float16)acc;
}

// ==================================================================
// x = LayerNorm(x + res) * g + be   (one block of 256 per row)
// also refreshes the f16 mirror xh used as GEMM A-operand
// ==================================================================
__global__ __launch_bounds__(256) void add_ln_kernel(
    float* __restrict__ x, _Float16* __restrict__ xh,
    const float* __restrict__ res, const float* __restrict__ g,
    const float* __restrict__ be) {
  __shared__ float s1[256];
  __shared__ float s2[256];
  const int row = blockIdx.x;
  const int t = threadIdx.x;
  float v = x[(size_t)row * DM + t] + res[(size_t)row * DM + t];
  s1[t] = v;
  s2[t] = v * v;
  __syncthreads();
  for (int s = 128; s > 0; s >>= 1) {
    if (t < s) {
      s1[t] += s1[t + s];
      s2[t] += s2[t + s];
    }
    __syncthreads();
  }
  float mu  = s1[0] * (1.0f / DM);
  float var = s2[0] * (1.0f / DM) - mu * mu;
  float inv = rsqrtf(var + 1e-5f);
  float o = (v - mu) * inv * g[t] + be[t];
  x[(size_t)row * DM + t]  = o;
  xh[(size_t)row * DM + t] = (_Float16)o;
}

// ==================================================================
// host-side orchestration
// ==================================================================
extern "C" void kernel_launch(void* const* d_in, const int* in_sizes, int n_in,
                              void* d_out, int out_size, void* d_ws,
                              size_t ws_size, hipStream_t stream) {
  (void)in_sizes; (void)n_in; (void)out_size; (void)ws_size;

  const float* src[4] = {(const float*)d_in[0], (const float*)d_in[2],
                         (const float*)d_in[4], (const float*)d_in[6]};
  const float* pos[4] = {(const float*)d_in[1], (const float*)d_in[3],
                         (const float*)d_in[5], (const float*)d_in[7]};
  const float* ref  = (const float*)d_in[8];
  const float* Woff = (const float*)d_in[9];
  const float* boff = (const float*)d_in[10];
  const float* Wa   = (const float*)d_in[11];
  const float* ba   = (const float*)d_in[12];
  const float* Wv   = (const float*)d_in[13];
  const float* bv   = (const float*)d_in[14];
  const float* Wo   = (const float*)d_in[15];
  const float* bo   = (const float*)d_in[16];
  const float* W1   = (const float*)d_in[17];
  const float* b1   = (const float*)d_in[18];
  const float* W2   = (const float*)d_in[19];
  const float* b2   = (const float*)d_in[20];
  const float* g1   = (const float*)d_in[21];
  const float* be1  = (const float*)d_in[22];
  const float* g2   = (const float*)d_in[23];
  const float* be2  = (const float*)d_in[24];

  // ---- workspace carve-up ----
  char* wsb = (char*)d_ws;
  size_t cur = 0;
  auto alloc = [&](size_t bytes) -> void* {
    void* p = wsb + cur;
    cur += (bytes + 255) & ~(size_t)255;
    return p;
  };
  float*    x     = (float*)alloc((size_t)MROWS * DM * 4);
  _Float16* xh    = (_Float16*)alloc((size_t)MROWS * DM * 2);
  float*    val   = (float*)alloc((size_t)MROWS * DM * 4);  // reused as tmp
  float*    offb  = (float*)alloc((size_t)MROWS * DM * 4);
  _Float16* msdah = (_Float16*)alloc((size_t)MROWS * DM * 2);
  float*    awb   = (float*)alloc((size_t)MROWS * (NHEADS * 16) * 4);
  _Float16* h1h   = (_Float16*)alloc((size_t)MROWS * DFFN * 2);
  _Float16* wt    = (_Float16*)alloc((size_t)DFFN * DM * 2);
  float*    tmp   = val;  // val is dead by the time tmp is produced

  // ---- build x = src_flat + pos (f32 + f16 mirror) ----
  const int HWs[4] = {10000, 2500, 625, 169};
  const int Ss[4]  = {0, 10000, 12500, 13125};
  for (int l = 0; l < 4; ++l) {
    int total = BB * HWs[l] * DM;
    build_x_kernel<<<(total + 255) / 256, 256, 0, stream>>>(src[l], pos[l], x,
                                                            xh, HWs[l], Ss[l]);
  }

  auto gemm = [&](const _Float16* A, const float* W, const float* bias,
                  float* Cf32, _Float16* Cf16, int M, int N, int K, int relu) {
    int tw = K * N;
    wconv_kernel<<<(tw + 255) / 256, 256, 0, stream>>>(W, wt, K, N);
    dim3 grid(N / 64, (M + 63) / 64);
    gemm_wmma_kernel<<<grid, 128, 0, stream>>>(A, wt, bias, Cf32, Cf16, M, N,
                                               K, relu);
  };

  for (int li = 0; li < NLAYERS; ++li) {
    // value / offset / attention-logit projections (f32 outputs)
    gemm(xh, Wv   + (size_t)li * DM * DM,  bv   + (size_t)li * DM,  val,  nullptr, MROWS, DM,  DM, 0);
    gemm(xh, Woff + (size_t)li * DM * DM,  boff + (size_t)li * DM,  offb, nullptr, MROWS, DM,  DM, 0);
    gemm(xh, Wa   + (size_t)li * DM * 128, ba   + (size_t)li * 128, awb,  nullptr, MROWS, 128, DM, 0);

    softmax16_kernel<<<(MROWS * NHEADS + 255) / 256, 256, 0, stream>>>(
        awb, MROWS * NHEADS);

    msda_sample_kernel<<<MROWS, 256, 0, stream>>>(val, offb, awb, ref, msdah);

    // output projection + residual + LN1
    gemm(msdah, Wo + (size_t)li * DM * DM, bo + (size_t)li * DM, tmp, nullptr, MROWS, DM, DM, 0);
    add_ln_kernel<<<MROWS, 256, 0, stream>>>(x, xh, tmp, g1 + (size_t)li * DM,
                                             be1 + (size_t)li * DM);

    // FFN (W1 output fused relu -> f16 hidden) + residual + LN2
    gemm(xh,  W1 + (size_t)li * DM * DFFN, b1 + (size_t)li * DFFN, nullptr, h1h, MROWS, DFFN, DM,   1);
    gemm(h1h, W2 + (size_t)li * DFFN * DM, b2 + (size_t)li * DM,   tmp, nullptr, MROWS, DM,   DFFN, 0);
    add_ln_kernel<<<MROWS, 256, 0, stream>>>(x, xh, tmp, g2 + (size_t)li * DM,
                                             be2 + (size_t)li * DM);
  }

  hipMemcpyAsync(d_out, x, (size_t)MROWS * DM * sizeof(float),
                 hipMemcpyDeviceToDevice, stream);
}
